// JointNetwork_2052994367587
// MI455X (gfx1250) — matrix operand order repaired and checked by reading, not compiled, not run on hardware
//
#include <hip/hip_runtime.h>
#include <hip/hip_bf16.h>
#include <math.h>

typedef _Float16 f16;
typedef _Float16 v16h __attribute__((ext_vector_type(16)));
typedef _Float16 v8h  __attribute__((ext_vector_type(8)));
typedef _Float16 v4h  __attribute__((ext_vector_type(4)));
typedef float    v8f  __attribute__((ext_vector_type(8)));

#define B_    8
#define T_    256
#define U_    64
#define Jdim  512
#define Vdim  1024
#define ZSTR  520   // padded LDS row stride (f16 elems): 1040B -> rows 4 banks apart

__device__ __forceinline__ float fast_tanh(float x) {
#if __has_builtin(__builtin_amdgcn_tanhf)
    return __builtin_amdgcn_tanhf(x);
#elif __has_builtin(__builtin_amdgcn_tanh_f32)
    return __builtin_amdgcn_tanh_f32(x);
#else
    return tanhf(x);
#endif
}

// ---------------------------------------------------------------------------
// Small fp32 projection GEMM: Y[r][j] = sum_k X[r][k] * W[j][k] (+ bias[j])
// X: R x 512, W: 512 x 512 row-major, Y: R x 512.  Grid: R/8 blocks, 256 thr.
// ---------------------------------------------------------------------------
__global__ __launch_bounds__(256)
void proj_kernel(const float* __restrict__ X, const float* __restrict__ W,
                 const float* __restrict__ bias, int has_bias,
                 float* __restrict__ Y) {
    __shared__ float xs[8][Jdim];  // 16 KB
    const int tid = threadIdx.x;
    const int r0  = blockIdx.x * 8;

    #pragma unroll
    for (int it = 0; it < 4; ++it) {
        int idx = it * 256 + tid;              // vec4 index over 8x128
        int rr  = idx >> 7;
        int c   = (idx & 127) * 4;
        *(float4*)&xs[rr][c] = *(const float4*)(X + (size_t)(r0 + rr) * Jdim + c);
    }
    __syncthreads();

    #pragma unroll
    for (int jj = 0; jj < 2; ++jj) {
        const int j = tid + jj * 256;
        const float* wr = W + (size_t)j * Jdim;
        float acc[8] = {0.f, 0.f, 0.f, 0.f, 0.f, 0.f, 0.f, 0.f};
        for (int k = 0; k < Jdim; k += 4) {
            float4 w4 = *(const float4*)(wr + k);
            #pragma unroll
            for (int r = 0; r < 8; ++r) {
                acc[r] = fmaf(w4.x, xs[r][k + 0],
                         fmaf(w4.y, xs[r][k + 1],
                         fmaf(w4.z, xs[r][k + 2],
                         fmaf(w4.w, xs[r][k + 3], acc[r]))));
            }
        }
        const float bv = has_bias ? bias[j] : 0.f;
        #pragma unroll
        for (int r = 0; r < 8; ++r)
            Y[(size_t)(r0 + r) * Jdim + j] = acc[r] + bv;
    }
}

// ---------------------------------------------------------------------------
// W_out fp32 -> f16 (row-major V x J).  Grid: 512 blocks x 256 thr, 4 elem/thr
// ---------------------------------------------------------------------------
__global__ __launch_bounds__(256)
void cvt_f16_kernel(const float* __restrict__ src, f16* __restrict__ dst) {
    const int i = (blockIdx.x * 256 + threadIdx.x) * 4;
    float4 v = *(const float4*)(src + i);
    v4h o;
    o.x = (f16)v.x; o.y = (f16)v.y; o.z = (f16)v.z; o.w = (f16)v.w;
    *(v4h*)(dst + i) = o;
}

// ---------------------------------------------------------------------------
// Main fused kernel: one block per (b,t).
//   Phase 1: z[u][j] = tanh(E[b,t][j] + D[b,u][j]) -> LDS (f16), 64 x 512
//   Phase 2: out[b,t,u,:] = z @ W_out^T + b_out via v_wmma_f32_16x16x32_f16
// 8 waves: wave = (mtile<<1)|nhalf ; each wave computes a 16(M) x 64(N) strip,
// looping over 8 N-chunks of 128 columns.
// ---------------------------------------------------------------------------
__global__ __launch_bounds__(256)
void joint_kernel(const float* __restrict__ E,    // (B*T) x J
                  const float* __restrict__ Dm,   // (B*U) x J
                  const f16*   __restrict__ W16,  // V x J  (f16)
                  const float* __restrict__ bout, // V
                  float*       __restrict__ out)  // (B*T*U) x V
{
    extern __shared__ char smem[];
    f16* zl = (f16*)smem;  // U_ rows x ZSTR f16

    const int tid = threadIdx.x;
    const int t   = blockIdx.x;
    const int b   = blockIdx.y;

    const float* e_row  = E  + (size_t)(b * T_ + t) * Jdim;
    const float* d_base = Dm + (size_t)b * U_ * Jdim;

    // Warm L2/L0 with the first 128-row W_out chunk (128 KB = 1024 lines)
    {
        const char* wb = (const char*)W16;
        #pragma unroll
        for (int p = 0; p < 4; ++p)
            __builtin_prefetch(wb + (size_t)(p * 256 + tid) * 128, 0, 1);
    }

    // ---- Phase 1: build z tile in LDS (f16) ----
    #pragma unroll 4
    for (int it = 0; it < 32; ++it) {
        int idx = it * 256 + tid;        // vec4 index over 64 x 128
        int u   = idx >> 7;
        int j   = (idx & 127) * 4;
        float4 dv = *(const float4*)(d_base + (size_t)u * Jdim + j);
        float4 ev = *(const float4*)(e_row + j);
        v4h o;
        o.x = (f16)fast_tanh(ev.x + dv.x);
        o.y = (f16)fast_tanh(ev.y + dv.y);
        o.z = (f16)fast_tanh(ev.z + dv.z);
        o.w = (f16)fast_tanh(ev.w + dv.w);
        *(v4h*)(zl + u * ZSTR + j) = o;
    }
    __syncthreads();

    // ---- Phase 2: WMMA ----
    const int lane  = tid & 31;
    const int lmod  = lane & 15;
    const int lhi   = lane >> 4;
    const int wave  = tid >> 5;
    const int mtile = wave >> 1;   // 0..3  -> rows u0 = mtile*16
    const int nhalf = wave & 1;    // 0..1  -> 64-column half of the 128 chunk

    // A-frag source row in LDS (16-bit A layout: lanes 0-15 rows, halves
    // [k..k+7] and [k+16..k+23]; lanes 16-31 get the +8 chunks)
    const f16* arow = zl + (size_t)(mtile * 16 + lmod) * ZSTR;
    const size_t outbase = (size_t)(b * T_ + t) * U_ * (size_t)Vdim;

    for (int nc = 0; nc < 8; ++nc) {
        const int n0 = nc * 128 + nhalf * 64;
        v8f acc[4];
        #pragma unroll
        for (int f = 0; f < 4; ++f) acc[f] = (v8f){0.f,0.f,0.f,0.f,0.f,0.f,0.f,0.f};

        for (int k0 = 0; k0 < Jdim; k0 += 32) {
            union { v16h v; v8h h[2]; } a;
            a.h[0] = *(const v8h*)(arow + k0 + lhi * 8);
            a.h[1] = *(const v8h*)(arow + k0 + 16 + lhi * 8);
            #pragma unroll
            for (int f = 0; f < 4; ++f) {
                // B layout (K x 16): lane = column n, lanes 0-15 hold K k0..k0+7 and
                // k0+16..k0+23, lanes 16-31 hold the +8 chunks (matches A)
                const f16* wp = W16 + (size_t)(n0 + f * 16 + lmod) * Jdim + k0 + lhi * 8;
                union { v16h v; v8h h[2]; } bm;
                bm.h[0] = *(const v8h*)(wp);
                bm.h[1] = *(const v8h*)(wp + 16);
                acc[f] = __builtin_amdgcn_wmma_f32_16x16x32_f16(
                    false, a.v, false, bm.v, (short)0, acc[f], false, false);
            }
        }

        // Epilogue: C layout -> lane holds column lmod, rows (8*lhi + i)
        #pragma unroll
        for (int f = 0; f < 4; ++f) {
            const int n  = n0 + f * 16 + lmod;
            const float bv = bout[n];
            const int u0 = mtile * 16 + lhi * 8;
            #pragma unroll
            for (int i = 0; i < 8; ++i)
                out[outbase + (size_t)(u0 + i) * Vdim + n] = acc[f][i] + bv;
        }
    }
}

// ---------------------------------------------------------------------------
extern "C" void kernel_launch(void* const* d_in, const int* in_sizes, int n_in,
                              void* d_out, int out_size, void* d_ws, size_t ws_size,
                              hipStream_t stream) {
    const float* h_enc = (const float*)d_in[0];  // 8*256*1*512
    const float* h_dec = (const float*)d_in[1];  // 8*1*64*512
    const float* W_enc = (const float*)d_in[2];  // 512*512
    const float* b_enc = (const float*)d_in[3];  // 512
    const float* W_dec = (const float*)d_in[4];  // 512*512
    const float* W_out = (const float*)d_in[5];  // 1024*512
    const float* b_out = (const float*)d_in[6];  // 1024
    float* out = (float*)d_out;

    char* ws = (char*)d_ws;
    float* Ew  = (float*)ws;                          // 2048*512 f32 = 4 MB
    float* Dw  = (float*)(ws + ((size_t)4 << 20));    //  512*512 f32 = 1 MB
    f16*   W16 = (f16*)  (ws + ((size_t)5 << 20));    // 1024*512 f16 = 1 MB

    // E = h_enc @ W_enc^T + b_enc   (2048 rows)
    proj_kernel<<<B_ * T_ / 8, 256, 0, stream>>>(h_enc, W_enc, b_enc, 1, Ew);
    // D = h_dec @ W_dec^T           (512 rows)
    proj_kernel<<<B_ * U_ / 8, 256, 0, stream>>>(h_dec, W_dec, nullptr, 0, Dw);
    // W_out -> f16
    cvt_f16_kernel<<<(Vdim * Jdim) / (256 * 4), 256, 0, stream>>>(W_out, W16);

    // Fused tanh + output GEMM
    dim3 grid(T_, B_);
    const size_t smem = (size_t)U_ * ZSTR * sizeof(f16);  // 66,560 B dynamic LDS
    joint_kernel<<<grid, 256, smem, stream>>>(Ew, Dw, W16, b_out, out);
}
